// ItemLayer_87814901334418
// MI455X (gfx1250) — matrix operand order repaired and checked by reading, not compiled, not run on hardware
//
#include <hip/hip_runtime.h>
#include <hip/hip_bf16.h>
#include <hip/hip_fp16.h>

#define N_OPS 100000
#define N_RES 1000
#define NEDGE 1600000
#define D_IN  128
#define D_HID 256
#define D_OUT 128

#define BM   32        // rows per block: two 16-row M-tiles
#define LDX  136       // 128 + 8 halves pad (bank rotate, keeps 16B align)
#define LDH  264       // 256 + 8

typedef __attribute__((ext_vector_type(16))) _Float16 v16h;
typedef __attribute__((ext_vector_type(8)))  _Float16 v8h;
typedef __attribute__((ext_vector_type(8)))  float    v8f;

struct MlpParams {
    const _Float16* wt[15];   // transposed f16 weights WT[n][k]; 0..11 small MLPs, 12..14 combined
    const float*    bias[15]; // f32 biases straight from inputs
};

// ---------------------------------------------------------------- utilities

__global__ __launch_bounds__(256) void zero_kernel(float* __restrict__ agg,
                                                   float* __restrict__ out, int n4) {
    int t = blockIdx.x * 256 + threadIdx.x;
    if (t < n4) {
        float4 z = {0.f, 0.f, 0.f, 0.f};
        ((float4*)agg)[t] = z;
        ((float4*)out)[t] = z;
    }
}

// dst[n*K + k] = (f16) src[k*N + n]   (W is [K][N] row-major from JAX)
__global__ __launch_bounds__(256) void transpose_cvt_kernel(const float* __restrict__ src,
                                                            _Float16* __restrict__ dst,
                                                            int K, int N) {
    int idx = blockIdx.x * 256 + threadIdx.x;
    if (idx < K * N) {
        int k = idx / N, n = idx - k * N;
        dst[(size_t)n * K + k] = (_Float16)src[idx];
    }
}

// segment-sum: agg[ops_idx[e]] += resources[res_idx[e]]  (128-wide rows)
__global__ __launch_bounds__(256) void scatter_kernel(const int* __restrict__ ops_idx,
                                                      const int* __restrict__ res_idx,
                                                      const float* __restrict__ resources,
                                                      float* __restrict__ agg) {
    long long t = (long long)blockIdx.x * 256 + threadIdx.x;
    int e = (int)(t >> 5);
    if (e >= NEDGE) return;
    int c = ((int)t & 31) << 2;                       // 32 lanes x 4 floats = 128
    int op = ops_idx[e], r = res_idx[e];
    const float4 v = *(const float4*)(resources + (size_t)r * D_OUT + c);
    float* dst = agg + (size_t)op * D_OUT + c;
    atomicAdd(dst + 0, v.x);
    atomicAdd(dst + 1, v.y);
    atomicAdd(dst + 2, v.z);
    atomicAdd(dst + 3, v.w);
}

// ------------------------------------------------------- WMMA fragment I/O
// A (16x32 f16): lanes 0-15 hold M=lane, halves K+0..15;
//                lanes 16-31 hold M=lane-16, halves K+16..31.
__device__ __forceinline__ v16h load_a(const _Float16* __restrict__ X, int ld,
                                       int row0, int k0, int m, int hi) {
    const _Float16* p = X + (row0 + m) * ld + k0 + hi * 16;
    union { v16h v; v8h h[2]; } u;
    u.h[0] = *(const v8h*)(p);
    u.h[1] = *(const v8h*)(p + 8);
    return u.v;
}

// B (32x16 f16): lane holds column N=lane&15; lanes 0-15 K=k0..15, 16-31 K=k0+16..31.
template<int STRIDE>
__device__ __forceinline__ v16h load_b(const _Float16* __restrict__ WT,
                                       int n0, int k0, int m, int hi) {
    return *(const v16h*)(WT + (size_t)(n0 + m) * STRIDE + k0 + hi * 16);
}

__device__ __forceinline__ v8f wmma_f16(v16h a, v16h b, v8f c) {
    return __builtin_amdgcn_wmma_f32_16x16x32_f16(false, a, false, b, (short)0, c, false, false);
}

// One layer over two M-tiles: [32 x K] @ WT[NOUT x K] + bias (+ELU) -> LDS f16.
// Each B fragment feeds both M-tiles (halves weight traffic).
template<bool ELU, int K, int LDI, int NOUT, int LDO>
__device__ __forceinline__ void layer2(const _Float16* __restrict__ Xin,
                                       const _Float16* __restrict__ WT,
                                       const float* __restrict__ Bv,
                                       _Float16* __restrict__ Hout,
                                       int wave, int lane) {
    const int m  = lane & 15;
    const int hi = (lane >> 4) & 1;
    for (int nt = wave; nt < (NOUT >> 4); nt += 8) {
        const int n0 = nt << 4;
        v8f a0 = {}, a1 = {};
#pragma unroll
        for (int k0 = 0; k0 < K; k0 += 32) {
            v16h b  = load_b<K>(WT, n0, k0, m, hi);
            v16h x0 = load_a(Xin, LDI, 0,  k0, m, hi);
            v16h x1 = load_a(Xin, LDI, 16, k0, m, hi);
            a0 = wmma_f16(x0, b, a0);
            a1 = wmma_f16(x1, b, a1);
        }
        const float bv = Bv[n0 + m];
#pragma unroll
        for (int j = 0; j < 8; ++j) {          // C/D: VGPR j -> M = j + hi*8, N = n0+m
            float v0 = a0[j] + bv, v1 = a1[j] + bv;
            if (ELU) {
                v0 = (v0 > 0.f) ? v0 : (__expf(v0) - 1.f);
                v1 = (v1 > 0.f) ? v1 : (__expf(v1) - 1.f);
            }
            Hout[(j + hi * 8) * LDO + n0 + m]      = (_Float16)v0;
            Hout[(16 + j + hi * 8) * LDO + n0 + m] = (_Float16)v1;
        }
    }
}

// Fold small-MLP output T[32x128] into persistent combined-L1 accumulators:
// accC[mt][t] += T @ W1c[kOff..kOff+127, n0..n0+15]   (WT stride 512)
__device__ __forceinline__ void fold_layer(const _Float16* __restrict__ Tin,
                                           const _Float16* __restrict__ WTc1,
                                           int kOff, v8f accC[2][2],
                                           int wave, int lane) {
    const int m  = lane & 15;
    const int hi = (lane >> 4) & 1;
#pragma unroll
    for (int t = 0; t < 2; ++t) {
        const int n0 = (wave + t * 8) << 4;
#pragma unroll
        for (int k0 = 0; k0 < 128; k0 += 32) {
            v16h b  = *(const v16h*)(WTc1 + (size_t)(n0 + m) * 512 + kOff + k0 + hi * 16);
            v16h x0 = load_a(Tin, LDX, 0,  k0, m, hi);
            v16h x1 = load_a(Tin, LDX, 16, k0, m, hi);
            accC[0][t] = wmma_f16(x0, b, accC[0][t]);
            accC[1][t] = wmma_f16(x1, b, accC[1][t]);
        }
    }
}

// Final combined layer: [32 x 256] @ WT[128 x 256] + bias -> f32 rows of d_out.
__device__ __forceinline__ void out_layer(const _Float16* __restrict__ Xin,
                                          const _Float16* __restrict__ WT,
                                          const float* __restrict__ Bv,
                                          float* __restrict__ out,
                                          int rowbase, int wave, int lane) {
    const int m  = lane & 15;
    const int hi = (lane >> 4) & 1;
    const int n0 = wave << 4;                 // NOUT=128: one N-tile per wave
    v8f a0 = {}, a1 = {};
#pragma unroll
    for (int k0 = 0; k0 < 256; k0 += 32) {
        v16h b  = load_b<256>(WT, n0, k0, m, hi);
        v16h x0 = load_a(Xin, LDH, 0,  k0, m, hi);
        v16h x1 = load_a(Xin, LDH, 16, k0, m, hi);
        a0 = wmma_f16(x0, b, a0);
        a1 = wmma_f16(x1, b, a1);
    }
    const float bv = Bv[n0 + m];
#pragma unroll
    for (int j = 0; j < 8; ++j) {
        int r0 = rowbase + j + hi * 8;
        int r1 = r0 + 16;
        if (r0 <= N_OPS - 2) out[(size_t)r0 * D_OUT + n0 + m] = a0[j] + bv;
        if (r1 <= N_OPS - 2) out[(size_t)r1 * D_OUT + n0 + m] = a1[j] + bv;
    }
}

// ------------------------------------------------------------- fused MLPs
__global__ __launch_bounds__(256) void mlp_kernel(const float* __restrict__ ops,
                                                  const float* __restrict__ agg,
                                                  const int* __restrict__ preds,
                                                  const int* __restrict__ succs,
                                                  float* __restrict__ out,
                                                  MlpParams P) {
    __shared__ _Float16 X[BM * LDX];      // per-MLP input; reused as small-MLP L3 out T
    __shared__ _Float16 H1[BM * LDH];
    __shared__ _Float16 H2[BM * LDH];
    __shared__ int gidx[3][BM];

    const int tid     = threadIdx.x;
    const int wave    = tid >> 5;
    const int lane    = tid & 31;
    const int rowbase = 1 + blockIdx.x * BM;

    if (tid < BM) {
        int row = rowbase + tid;
        if (row > N_OPS - 2) row = N_OPS - 2;   // clamp; final stores are row-guarded
        gidx[0][tid] = preds[row];
        gidx[1][tid] = succs[row];
        gidx[2][tid] = row;
    }

    v8f accC[2][2] = {};   // persistent combined-L1 accumulators (this wave's N-tiles)

    // Four small MLPs: 128 -ELU-> 256 -ELU-> 256 -> 128, folded into accC
#pragma unroll 1
    for (int m = 0; m < 4; ++m) {
        __syncthreads();   // prior T reads (fold) / gidx writes complete
        const float* src = (m == 2) ? agg : ops;
        const int*   ia  = (m == 0) ? gidx[0] : (m == 1) ? gidx[1] : gidx[2];
        for (int idx = tid; idx < BM * D_IN; idx += 256) {
            int i = idx >> 7, c = idx & 127;
            X[i * LDX + c] = (_Float16)src[(size_t)ia[i] * D_IN + c];
        }
        __syncthreads();
        layer2<true , 128, LDX, 256, LDH>(X,  P.wt[m*3+0], P.bias[m*3+0], H1, wave, lane);
        __syncthreads();
        layer2<true , 256, LDH, 256, LDH>(H1, P.wt[m*3+1], P.bias[m*3+1], H2, wave, lane);
        __syncthreads();
        layer2<false, 256, LDH, 128, LDX>(H2, P.wt[m*3+2], P.bias[m*3+2], X,  wave, lane); // T
        __syncthreads();
        fold_layer(X, P.wt[12], m * 128, accC, wave, lane);
    }
    __syncthreads();

    // Combined L1 epilogue: bias + ELU on register accumulators -> H1 (f16)
    {
        const float* Bv = P.bias[12];
        const int m  = lane & 15;
        const int hi = (lane >> 4) & 1;
#pragma unroll
        for (int t = 0; t < 2; ++t) {
            const int n0 = (wave + t * 8) << 4;
            const float bv = Bv[n0 + m];
#pragma unroll
            for (int mt = 0; mt < 2; ++mt) {
#pragma unroll
                for (int j = 0; j < 8; ++j) {
                    float v = accC[mt][t][j] + bv;
                    v = (v > 0.f) ? v : (__expf(v) - 1.f);
                    H1[(mt * 16 + j + hi * 8) * LDH + n0 + m] = (_Float16)v;
                }
            }
        }
    }
    __syncthreads();
    layer2<true, 256, LDH, 256, LDH>(H1, P.wt[13], P.bias[13], H2, wave, lane);
    __syncthreads();
    out_layer(H2, P.wt[14], P.bias[14], out, rowbase, wave, lane);
}

// ------------------------------------------------------------------ launch
extern "C" void kernel_launch(void* const* d_in, const int* in_sizes, int n_in,
                              void* d_out, int out_size, void* d_ws, size_t ws_size,
                              hipStream_t stream) {
    (void)in_sizes; (void)n_in; (void)out_size; (void)ws_size;

    const float* ops       = (const float*)d_in[0];
    const float* resources = (const float*)d_in[1];
    const int*   edges     = (const int*)d_in[2];
    const int*   ops_idx   = edges;               // stack([ops_idx, res_idx]) flat
    const int*   res_idx   = edges + NEDGE;
    const int*   preds     = (const int*)d_in[3];
    const int*   succs     = (const int*)d_in[4];
    float*       out       = (float*)d_out;

    // d_ws: [agg f32 N_OPS x 128][f16 transposed weights, 15 arrays]
    float* agg = (float*)d_ws;
    size_t off = (size_t)N_OPS * D_IN * sizeof(float);

    // params flattened in setup_inputs() insertion order:
    // combined @5, predecessor @11, successor @17, resources @23, same @29
    const int baseIdx[5] = {11, 17, 23, 29, 5};   // my order: pred,succ,res,same,comb
    const int K1[5]      = {128, 128, 128, 128, 512};

    {   // zero agg + out
        int n4 = N_OPS * D_IN / 4;
        zero_kernel<<<(n4 + 255) / 256, 256, 0, stream>>>(agg, out, n4);
    }

    MlpParams P;
    for (int m = 0; m < 5; ++m) {
        for (int l = 0; l < 3; ++l) {
            int K = (l == 0) ? K1[m] : D_HID;
            int N = (l == 2) ? D_OUT : D_HID;
            _Float16* dst = (_Float16*)((char*)d_ws + off);
            off += (size_t)K * N * sizeof(_Float16);
            const float* wsrc = (const float*)d_in[baseIdx[m] + 2 * l];
            int total = K * N;
            transpose_cvt_kernel<<<(total + 255) / 256, 256, 0, stream>>>(wsrc, dst, K, N);
            P.wt[m * 3 + l]   = dst;
            P.bias[m * 3 + l] = (const float*)d_in[baseIdx[m] + 2 * l + 1];
        }
    }

    {   // edge scatter-add (atomic segment sum)
        long long tot = (long long)NEDGE * 32;
        int blocks = (int)((tot + 255) / 256);
        scatter_kernel<<<blocks, 256, 0, stream>>>(ops_idx, res_idx, resources, agg);
    }

    {   // fused five-MLP kernel, 32 rows per 8-wave block
        int nblocks = (N_OPS - 2 + BM - 1) / BM;
        mlp_kernel<<<nblocks, 256, 0, stream>>>(ops, agg, preds, succs, out, P);
    }
}
